// MultiHeadAttention_86620900426395
// MI455X (gfx1250) — matrix operand order repaired and checked by reading, not compiled, not run on hardware
//
#include <hip/hip_runtime.h>
#include <stdint.h>

#define D_MODEL 1024
#define N_HEADS 16
#define HEAD_DIM 64
#define SEQ 2048
#define BATCH 2

typedef __attribute__((ext_vector_type(16))) __bf16 bf16x16;
typedef __attribute__((ext_vector_type(8))) float f32x8;
typedef __attribute__((ext_vector_type(4))) unsigned int u32x4;
typedef __attribute__((ext_vector_type(8))) int i32x8;
typedef __attribute__((ext_vector_type(4))) int i32x4;

union Frag { bf16x16 v; uint32_t u[8]; };

__device__ __forceinline__ uint16_t f32_to_bf16(float f) {
  uint32_t u = __float_as_uint(f);
  uint32_t r = (u + 0x7FFFu + ((u >> 16) & 1u)) >> 16;  // round-to-nearest-even
  return (uint16_t)r;
}

// ---------------------------------------------------------------------------
// Tensor Data Mover: DMA a 2-D bf16 tile (tile_x contiguous elems per row,
// tile_y rows, row stride = stride_x elems) from global into LDS at lds_off.
// D# per CDNA5 ISA ch.8; descriptors live in SGPRs; tracked by TENSORcnt.
// ---------------------------------------------------------------------------
__device__ __forceinline__ void tdm_load_2d(uint32_t lds_off, const void* gaddr,
                                            uint32_t tile_x, uint32_t tile_y,
                                            uint64_t stride_x_elems) {
  uint64_t ga = (uint64_t)(uintptr_t)gaddr;
  u32x4 g0;
  g0[0] = 1u;                                   // count=1 (valid descriptor)
  g0[1] = lds_off;                              // LDS byte address
  g0[2] = (uint32_t)ga;                         // global_addr[31:0]
  g0[3] = (uint32_t)((ga >> 32) & 0x01FFFFFFu)  // global_addr[56:32]
          | (2u << 30);                         // type = 2 ("image")
  uint32_t tx = tile_x, ty = tile_y;            // tensor dims == tile dims
  uint64_t s0 = stride_x_elems;
  i32x8 g1;
  g1[0] = (int)(1u << 16);                      // data_size = 1 -> 2 bytes
  g1[1] = (int)((tx & 0xFFFFu) << 16);          // tensor_dim0[15:0] @ bits 63:48
  g1[2] = (int)((tx >> 16) | ((ty & 0xFFFFu) << 16));   // dim0 hi | dim1 lo
  g1[3] = (int)((ty >> 16) | (tile_x << 16));   // dim1 hi | tile_dim0
  g1[4] = (int)tile_y;                          // tile_dim1 | tile_dim2=0
  g1[5] = (int)(uint32_t)s0;                    // tensor_dim0_stride[31:0]
  g1[6] = (int)((uint32_t)(s0 >> 32) & 0xFFFFu);// stride0[47:32] | stride1 lo=0
  g1[7] = 0;
  i32x4 z4 = {};
#if __clang_major__ >= 23
  i32x8 z8 = {};
  __builtin_amdgcn_tensor_load_to_lds(g0, g1, z4, z4, z8, 0);
#else
  __builtin_amdgcn_tensor_load_to_lds(g0, g1, z4, z4, 0);
#endif
}

// ---------------------------------------------------------------------------
// f32 -> bf16, 4 elems/thread (float4 load, 64-bit packed store)
// ---------------------------------------------------------------------------
__global__ void cvt_f32_to_bf16(const float4* __restrict__ in,
                                uint64_t* __restrict__ out, int n4) {
  int i = blockIdx.x * blockDim.x + threadIdx.x;
  int stride = gridDim.x * blockDim.x;
  for (; i < n4; i += stride) {
    float4 f = in[i];
    uint64_t p = (uint64_t)f32_to_bf16(f.x) | ((uint64_t)f32_to_bf16(f.y) << 16) |
                 ((uint64_t)f32_to_bf16(f.z) << 32) | ((uint64_t)f32_to_bf16(f.w) << 48);
    out[i] = p;
  }
}

// ---------------------------------------------------------------------------
// Weight convert + transpose: W [K][N] f32 -> Wt [N][K] bf16.
// Makes the WMMA B-operand K-dim contiguous (b128 LDS fragment loads).
// ---------------------------------------------------------------------------
__global__ void cvt_f32_to_bf16_T(const float* __restrict__ in,
                                  uint16_t* __restrict__ out,
                                  int Kdim, int Ndim) {
  int i = blockIdx.x * blockDim.x + threadIdx.x;
  int stride = gridDim.x * blockDim.x;
  int total = Kdim * Ndim;
  for (; i < total; i += stride) {
    int k = i / Ndim, n = i - k * Ndim;  // coalesced read along N
    out[(size_t)n * Kdim + k] = f32_to_bf16(in[i]);
  }
}

// ---------------------------------------------------------------------------
// C[M,N] = A[M,K](bf16) * Wt[N,K](bf16, pre-transposed) + bias (f32).
// 128 threads (4 waves), block tile 64x64, wave tile 32x32, K-step 32.
// TDM double-buffered staging: A tile [row][k] 64x32, B tile [n][k] 64x32.
// outMode: 0 = bf16 row-major, 1 = f32 row-major, 2 = bf16 V-transposed
//          (Vt [B,H,Hd,T] so attention's V B-operand has contiguous keys).
// ---------------------------------------------------------------------------
__global__ __launch_bounds__(128)
void gemm_bias(const uint16_t* __restrict__ A, const uint16_t* __restrict__ Wt,
               const float* __restrict__ bias, void* __restrict__ out,
               int M, int N, int K, int outMode) {
  __shared__ __align__(16) uint16_t ldsA[2][64 * 32];
  __shared__ __align__(16) uint16_t ldsB[2][64 * 32];  // [n][k]

  int tid = threadIdx.x;
  int lane = tid & 31;
  int half = lane >> 4, l16 = lane & 15;
  int w = tid >> 5;
  int wr = w >> 1, wc = w & 1;
  int m0 = blockIdx.y * 64, n0 = blockIdx.x * 64;

  uint32_t aOff[2] = {(uint32_t)(uintptr_t)&ldsA[0][0], (uint32_t)(uintptr_t)&ldsA[1][0]};
  uint32_t bOff[2] = {(uint32_t)(uintptr_t)&ldsB[0][0], (uint32_t)(uintptr_t)&ldsB[1][0]};

  if (tid == 0) {  // prologue DMA into buffer 0
    tdm_load_2d(aOff[0], A + (size_t)m0 * K, 32, 64, (uint64_t)K);
    tdm_load_2d(bOff[0], Wt + (size_t)n0 * K, 32, 64, (uint64_t)K);
  }

  f32x8 acc[2][2] = {};
  for (int k0 = 0; k0 < K; k0 += 32) {
    int cur = (k0 >> 5) & 1;
    bool hasNext = (k0 + 32) < K;
    if (hasNext && tid == 0) {  // prefetch next K-slice into the other buffer
      tdm_load_2d(aOff[cur ^ 1], A + (size_t)m0 * K + (k0 + 32), 32, 64, (uint64_t)K);
      tdm_load_2d(bOff[cur ^ 1], Wt + (size_t)n0 * K + (k0 + 32), 32, 64, (uint64_t)K);
    }
    if (hasNext) __builtin_amdgcn_s_wait_tensorcnt(2);  // current 2 DMAs done
    else         __builtin_amdgcn_s_wait_tensorcnt(0);
    __syncthreads();

    const uint32_t* lA32 = (const uint32_t*)&ldsA[cur][0];
    const uint32_t* lB32 = (const uint32_t*)&ldsB[cur][0];

    Frag a[2], b[2];
#pragma unroll
    for (int mi = 0; mi < 2; ++mi) {
      int row = wr * 32 + mi * 16 + l16;  // A layout: lane = row
#pragma unroll
      for (int v = 0; v < 8; ++v) {
        int kd = (v & 3) + half * 4 + (v >> 2) * 8;  // K dword idx (16-bit A 16x32)
        a[mi].u[v] = lA32[row * 16 + kd];
      }
    }
#pragma unroll
    for (int ni = 0; ni < 2; ++ni) {
      int col = wc * 32 + ni * 16 + l16;  // B layout: lane = col
#pragma unroll
      for (int v = 0; v < 8; ++v)
        b[ni].u[v] = lB32[col * 16 + (v & 3) + half * 4 + (v >> 2) * 8];  // K pair contiguous in [n][k]
    }
#pragma unroll
    for (int mi = 0; mi < 2; ++mi)
#pragma unroll
      for (int ni = 0; ni < 2; ++ni)
        acc[mi][ni] = __builtin_amdgcn_wmma_f32_16x16x32_bf16(
            false, a[mi].v, false, b[ni].v, (short)0, acc[mi][ni], false, false);
    __syncthreads();  // all waves done reading before buffer is re-DMAed
  }

  // epilogue: C layout m = v + half*8, n = l16
#pragma unroll
  for (int mi = 0; mi < 2; ++mi)
#pragma unroll
    for (int ni = 0; ni < 2; ++ni)
#pragma unroll
      for (int v = 0; v < 8; ++v) {
        int gr = m0 + wr * 32 + mi * 16 + half * 8 + v;
        int gc = n0 + wc * 32 + ni * 16 + l16;
        float val = acc[mi][ni][v] + bias[gc];
        if (outMode == 1) {
          ((float*)out)[(size_t)gr * N + gc] = val;
        } else if (outMode == 2) {
          // Vt[b, h, d, t]: (b*1024 + gc) rows of length SEQ, key index = t
          size_t idx = ((size_t)(gr >> 11) * D_MODEL + gc) * SEQ + (gr & (SEQ - 1));
          ((uint16_t*)out)[idx] = f32_to_bf16(val);
        } else {
          ((uint16_t*)out)[(size_t)gr * N + gc] = f32_to_bf16(val);
        }
      }
}

// ---------------------------------------------------------------------------
// Causal flash attention. Q/K: [B*T, 1024] bf16 row-major; V: Vt [B,H,Hd,T].
// Grid (T/64, B*H), 128 threads; wave w owns q-rows qb+w*16..+15.
// K tiles [key][d] and Vt tiles [d][key] staged via double-buffered TDM.
// ---------------------------------------------------------------------------
__global__ __launch_bounds__(128)
void flash_attn(const uint16_t* __restrict__ Qb, const uint16_t* __restrict__ Kb,
                const uint16_t* __restrict__ Vt, uint16_t* __restrict__ Ob) {
  __shared__ __align__(16) uint16_t ldsK[2][64 * 64];   // [key][d]
  __shared__ __align__(16) uint16_t ldsVt[2][64 * 64];  // [d][key]
  __shared__ __align__(16) uint16_t ldsP[4 * 16 * 64];  // per-wave 16x64 P tile

  int tid = threadIdx.x;
  int lane = tid & 31, w = tid >> 5;
  int half = lane >> 4, l16 = lane & 15;
  int bh = blockIdx.y;
  int b = bh >> 4, h = bh & 15;
  int qb = blockIdx.x * 64;

  const uint32_t* Q32 = (const uint32_t*)Qb;
  size_t rowBase = (size_t)b * SEQ;
  int hc = h * HEAD_DIM;
  const uint16_t* VtHead = Vt + (size_t)bh * HEAD_DIM * SEQ;  // [d][t]

  uint32_t kOff[2] = {(uint32_t)(uintptr_t)&ldsK[0][0], (uint32_t)(uintptr_t)&ldsK[1][0]};
  uint32_t vOff[2] = {(uint32_t)(uintptr_t)&ldsVt[0][0], (uint32_t)(uintptr_t)&ldsVt[1][0]};

  // Q fragments (A layout), held in registers for the whole kernel
  Frag qf[2];
  {
    size_t qrow = rowBase + qb + w * 16 + l16;
#pragma unroll
    for (int s = 0; s < 2; ++s)
#pragma unroll
      for (int v = 0; v < 8; ++v) {
        int d = (v & 3) * 2 + half * 8 + (v >> 2) * 16 + s * 32;
        qf[s].u[v] = Q32[(qrow * D_MODEL + hc + d) >> 1];
      }
  }

  f32x8 accO[4] = {};
  float rowm[8], rowl[8];
#pragma unroll
  for (int v = 0; v < 8; ++v) { rowm[v] = -1e30f; rowl[v] = 0.0f; }

  uint16_t* myP = ldsP + w * (16 * 64);
  const uint32_t* myP32 = (const uint32_t*)myP;

  int nblocks = (qb >> 6) + 1;
  if (tid == 0) {  // prologue DMA: key-block 0 into buffer 0
    tdm_load_2d(kOff[0], Kb + rowBase * D_MODEL + hc, 64, 64, (uint64_t)D_MODEL);
    tdm_load_2d(vOff[0], VtHead, 64, 64, (uint64_t)SEQ);
  }

  for (int ib = 0; ib < nblocks; ++ib) {
    int k0 = ib * 64;
    int cur = ib & 1;
    bool hasNext = (ib + 1) < nblocks;
    if (hasNext && tid == 0) {  // prefetch next key-block
      tdm_load_2d(kOff[cur ^ 1], Kb + (rowBase + k0 + 64) * D_MODEL + hc, 64, 64,
                  (uint64_t)D_MODEL);
      tdm_load_2d(vOff[cur ^ 1], VtHead + (k0 + 64), 64, 64, (uint64_t)SEQ);
    }
    if (hasNext) __builtin_amdgcn_s_wait_tensorcnt(2);
    else         __builtin_amdgcn_s_wait_tensorcnt(0);
    __syncthreads();

    const uint32_t* lK32 = (const uint32_t*)&ldsK[cur][0];
    const uint32_t* lVt32 = (const uint32_t*)&ldsVt[cur][0];

    // ---- S = Q * K^T : 16 x 64 scores per wave -------------------------
    f32x8 st[4];
#pragma unroll
    for (int t = 0; t < 4; ++t) {
      f32x8 s = {};
#pragma unroll
      for (int dstep = 0; dstep < 2; ++dstep) {
        Frag kf;  // B fragment: lane = key col; K-dim = head dim (contiguous)
        int keyl = t * 16 + l16;
#pragma unroll
        for (int v = 0; v < 8; ++v) {
          int d = dstep * 32 + (v & 3) * 2 + half * 8 + (v >> 2) * 16;
          kf.u[v] = lK32[(keyl << 5) + (d >> 1)];
        }
        s = __builtin_amdgcn_wmma_f32_16x16x32_bf16(
            false, qf[dstep].v, false, kf.v, (short)0, s, false, false);
      }
      st[t] = s;
    }
    // ---- scale + causal mask (C layout: row = half*8+v, col = l16) -----
#pragma unroll
    for (int t = 0; t < 4; ++t)
#pragma unroll
      for (int v = 0; v < 8; ++v) {
        int key = k0 + t * 16 + l16;
        int qq = qb + w * 16 + half * 8 + v;
        float sv = st[t][v] * 0.125f;  // 1/sqrt(64)
        st[t][v] = (key > qq) ? -1e30f : sv;
      }
    // ---- online softmax, row-wise (16-lane groups on wave32) -----------
#pragma unroll
    for (int v = 0; v < 8; ++v) {
      float mx = fmaxf(fmaxf(st[0][v], st[1][v]), fmaxf(st[2][v], st[3][v]));
#pragma unroll
      for (int off = 1; off < 16; off <<= 1)
        mx = fmaxf(mx, __shfl_xor(mx, off, 32));
      float mnew = fmaxf(rowm[v], mx);
      float alpha = __expf(rowm[v] - mnew);
      float ps = 0.0f;
      int prow = half * 8 + v;
#pragma unroll
      for (int t = 0; t < 4; ++t) {
        float p = __expf(st[t][v] - mnew);
        ps += p;
        myP[prow * 64 + t * 16 + l16] = f32_to_bf16(p);
      }
#pragma unroll
      for (int off = 1; off < 16; off <<= 1)
        ps += __shfl_xor(ps, off, 32);
      rowl[v] = rowl[v] * alpha + ps;
      rowm[v] = mnew;
#pragma unroll
      for (int t = 0; t < 4; ++t) accO[t][v] *= alpha;
    }
    // ---- O += P * V  (P via LDS in A layout; V from Vt tile, key pairs
    //      contiguous -> single dword LDS loads) -------------------------
#pragma unroll
    for (int dstep = 0; dstep < 2; ++dstep) {
      Frag pf;
#pragma unroll
      for (int v = 0; v < 8; ++v) {
        int kk = (v & 3) * 2 + half * 8 + (v >> 2) * 16 + dstep * 32;
        pf.u[v] = myP32[(l16 * 64 + kk) >> 1];
      }
#pragma unroll
      for (int t = 0; t < 4; ++t) {
        Frag vf;  // B fragment: lane = head-dim col; K-dim = key rows
        int n = t * 16 + l16;
#pragma unroll
        for (int v = 0; v < 8; ++v) {
          int kv = dstep * 32 + (v & 3) * 2 + half * 8 + (v >> 2) * 16;
          vf.u[v] = lVt32[(n << 5) + (kv >> 1)];
        }
        accO[t] = __builtin_amdgcn_wmma_f32_16x16x32_bf16(
            false, pf.v, false, vf.v, (short)0, accO[t], false, false);
      }
    }
    __syncthreads();  // done with ldsK/ldsVt[cur] before it is re-DMAed
  }
  // ---- normalize + store bf16 ------------------------------------------
#pragma unroll
  for (int t = 0; t < 4; ++t)
#pragma unroll
    for (int v = 0; v < 8; ++v) {
      int qq = qb + w * 16 + half * 8 + v;
      float o = accO[t][v] / rowl[v];
      Ob[(rowBase + qq) * D_MODEL + hc + t * 16 + l16] = f32_to_bf16(o);
    }
}

// ---------------------------------------------------------------------------
extern "C" void kernel_launch(void* const* d_in, const int* in_sizes, int n_in,
                              void* d_out, int out_size, void* d_ws, size_t ws_size,
                              hipStream_t stream) {
  const float* query = (const float*)d_in[0];
  const float* key   = (const float*)d_in[1];
  const float* value = (const float*)d_in[2];
  const float* Wq = (const float*)d_in[3];
  const float* bq = (const float*)d_in[4];
  const float* Wk = (const float*)d_in[5];
  const float* bk = (const float*)d_in[6];
  const float* Wv = (const float*)d_in[7];
  const float* bv = (const float*)d_in[8];
  const float* Wo = (const float*)d_in[9];
  const float* bo = (const float*)d_in[10];

  const size_t MT  = (size_t)BATCH * SEQ;      // 4096 rows
  const size_t ACT = MT * D_MODEL;             // 4M elems
  const size_t WEL = (size_t)D_MODEL * D_MODEL;

  uint16_t* xq  = (uint16_t*)d_ws;
  uint16_t* xk  = xq + ACT;
  uint16_t* xv  = xk + ACT;
  uint16_t* wqt = xv + ACT;   // transposed bf16 weights [N][K]
  uint16_t* wkt = wqt + WEL;
  uint16_t* wvt = wkt + WEL;
  uint16_t* wot = wvt + WEL;
  uint16_t* Qb  = wot + WEL;
  uint16_t* Kb  = Qb + ACT;
  uint16_t* Vt  = Kb + ACT;   // V stored transposed per head [B,H,Hd,T]
  uint16_t* Ab  = Vt + ACT;

  // 1) precision conversion (activations) + convert-transpose (weights)
  cvt_f32_to_bf16<<<dim3(1024), dim3(256), 0, stream>>>((const float4*)query, (uint64_t*)xq, (int)(ACT / 4));
  cvt_f32_to_bf16<<<dim3(1024), dim3(256), 0, stream>>>((const float4*)key,   (uint64_t*)xk, (int)(ACT / 4));
  cvt_f32_to_bf16<<<dim3(1024), dim3(256), 0, stream>>>((const float4*)value, (uint64_t*)xv, (int)(ACT / 4));
  cvt_f32_to_bf16_T<<<dim3(1024), dim3(256), 0, stream>>>(Wq, wqt, D_MODEL, D_MODEL);
  cvt_f32_to_bf16_T<<<dim3(1024), dim3(256), 0, stream>>>(Wk, wkt, D_MODEL, D_MODEL);
  cvt_f32_to_bf16_T<<<dim3(1024), dim3(256), 0, stream>>>(Wv, wvt, D_MODEL, D_MODEL);
  cvt_f32_to_bf16_T<<<dim3(1024), dim3(256), 0, stream>>>(Wo, wot, D_MODEL, D_MODEL);

  // 2) Q/K/V projections (WMMA GEMM + TDM staging)
  dim3 gblk(128), ggrd(D_MODEL / 64, (unsigned)(MT / 64));
  gemm_bias<<<ggrd, gblk, 0, stream>>>(xq, wqt, bq, Qb, (int)MT, D_MODEL, D_MODEL, 0);
  gemm_bias<<<ggrd, gblk, 0, stream>>>(xk, wkt, bk, Kb, (int)MT, D_MODEL, D_MODEL, 0);
  gemm_bias<<<ggrd, gblk, 0, stream>>>(xv, wvt, bv, Vt, (int)MT, D_MODEL, D_MODEL, 2);

  // 3) causal flash attention (WMMA + double-buffered TDM staging)
  flash_attn<<<dim3(SEQ / 64, BATCH * N_HEADS), gblk, 0, stream>>>(Qb, Kb, Vt, Ab);

  // 4) output projection (WMMA GEMM, f32 out + bias)
  gemm_bias<<<ggrd, gblk, 0, stream>>>(Ab, wot, bo, (float*)d_out,
                                       (int)MT, D_MODEL, D_MODEL, 1);
}
